// Hawkes_74938589381026
// MI455X (gfx1250) — matrix operand order repaired and checked by reading, not compiled, not run on hardware
//
#include <hip/hip_runtime.h>
#include <hip/hip_bf16.h>
#include <hip/hip_fp16.h>

// ---------------------------------------------------------------------------
// Hawkes intensity kernel for MI455X (gfx1250, wave32).
//
// excite[b,j,k] = sum_{i<=j} alpha[b,i,k] * exp(-delta[b,i,k]*(ts[j+1]-ts[i]))
//
//  * alpha/delta MLP input is a concat of one-hots -> only 4*3*4 = 48 distinct
//    rows. Computed once with V_WMMA_F32_16X16X32_F16 (f32 accumulate).
//  * mu MLP has 4 distinct rows (eye4) -> 4-entry table.
//  * O(I^2) excitation loop is v_exp_f32/VALU bound (~17M terms); alpha+delta
//    packed as float2 in LDS -> one ds_load_b64 per term; full tiles unrolled.
// ---------------------------------------------------------------------------

typedef __attribute__((ext_vector_type(16))) _Float16 v16h;
typedef __attribute__((ext_vector_type(8)))  float    v8f;

#define Bsz 8
#define Tsz 1024
#define Isz 1023
#define OFF_MU    32736
#define OFF_ALPHA 65504
#define OFF_DELTA 98240
#define TILE_I 128

// ---------------------------------------------------------------------------
// Kernel 1: per-row inclusive prefix sum (8 rows of 1024), one wave32 per row.
// ---------------------------------------------------------------------------
__global__ void hk_cumsum_kernel(const float* __restrict__ time_step,
                                 float* __restrict__ ts) {
    const int wave = threadIdx.x >> 5;   // 8 waves = 8 batch rows
    const int lane = threadIdx.x & 31;
    const int b = wave;
    float running = 0.f;
    for (int c = 0; c < Tsz / 32; ++c) {
        const int idx = c * 32 + lane;
        float v = (idx == 0) ? 0.f : time_step[b * Tsz + idx];
        #pragma unroll
        for (int off = 1; off < 32; off <<= 1) {
            float n = __shfl_up(v, off, 32);
            if (lane >= off) v += n;
        }
        v += running;
        ts[b * Tsz + idx] = v;
        running = __shfl(v, 31, 32);
    }
}

// ---------------------------------------------------------------------------
// Kernel 2: 48-entry alpha/delta tables + 4-entry mu table via WMMA.
// One wave per 16-row tile: roles 0-2 alpha, 3-5 delta, 6 mu.
// ---------------------------------------------------------------------------
__device__ __forceinline__ float hk_softplus(float x) {
    return fmaxf(x, 0.f) + log1pf(__expf(-fabsf(x)));
}

__global__ void hk_tables_kernel(const float* __restrict__ mu_w1, const float* __restrict__ mu_b1,
                                 const float* __restrict__ mu_w2, const float* __restrict__ mu_b2,
                                 const float* __restrict__ a_w1,  const float* __restrict__ a_b1,
                                 const float* __restrict__ a_w2,  const float* __restrict__ a_b2,
                                 const float* __restrict__ d_w1,  const float* __restrict__ d_b1,
                                 const float* __restrict__ d_w2,  const float* __restrict__ d_b2,
                                 float* __restrict__ a_tab, float* __restrict__ d_tab,
                                 float* __restrict__ mu_base) {
    const int role = blockIdx.x;          // 0..6
    const int L    = threadIdx.x;         // 0..31
    const int half = L >> 4;
    const int lm   = L & 15;

    const float *w1, *b1, *w2, *b2;
    float* tab;
    int tile, Kdim;
    if (role < 3)      { w1=a_w1;  b1=a_b1;  w2=a_w2;  b2=a_b2;  tab=a_tab;   tile=role;   Kdim=11; }
    else if (role < 6) { w1=d_w1;  b1=d_b1;  w2=d_w2;  b2=d_b2;  tab=d_tab;   tile=role-3; Kdim=11; }
    else               { w1=mu_w1; b1=mu_b1; w2=mu_w2; b2=mu_b2; tab=mu_base; tile=0;      Kdim=4;  }

    // One-hot input row for row (tile*16 + lm).
    int x[16];
    #pragma unroll
    for (int k = 0; k < 16; ++k) x[k] = 0;
    if (role < 6) {
        const int r  = tile * 16 + lm;      // 0..47
        const int ev = r / 12;
        const int st = (r % 12) / 4;
        const int kk = r % 4;
        x[ev] = 1; x[4 + st] = 1; x[7 + kk] = 1;
    } else {
        x[lm] = 1;                          // eye4 rows 0..3 (rest hit zero W1 rows)
    }

    // A fragment: 16-bit A 16x32 layout (ISA 7.12.2):
    //   jj<8  -> k = jj + 8*half ; jj>=8 -> k = 16 + (jj-8) + 8*half
    v16h a;
    #pragma unroll
    for (int jj = 0; jj < 16; ++jj) {
        const int kA = (jj < 8) ? (jj + 8 * half) : (16 + (jj - 8) + 8 * half);
        a[jj] = (kA < 16 && x[kA]) ? (_Float16)1.0f : (_Float16)0.0f;
    }

    // B fragment: 16-bit B 32x16, same K order as A:
    //   jj<8 -> k = jj + 8*half ; jj>=8 -> k = 16 + (jj-8) + 8*half, column n = lm
    const int n = lm;
    v16h bfr;
    #pragma unroll
    for (int jj = 0; jj < 16; ++jj) {
        const int kB = (jj < 8) ? (jj + 8 * half) : (16 + (jj - 8) + 8 * half);
        bfr[jj] = (kB < Kdim) ? (_Float16)w1[kB * 16 + n] : (_Float16)0.0f;
    }

    v8f c = {0.f, 0.f, 0.f, 0.f, 0.f, 0.f, 0.f, 0.f};
    c = __builtin_amdgcn_wmma_f32_16x16x32_f16(false, a, false, bfr,
                                               (short)0, c, false, false);

    // Second layer: out[m] = softplus( sum_n relu(c[m,n]+b1[n]) * w2[n] + b2 )
    const float b1n = b1[n];
    const float w2n = w2[n];
    const float b2v = b2[0];
    float red[8];
    #pragma unroll
    for (int r8 = 0; r8 < 8; ++r8) {
        float v = fmaxf(c[r8] + b1n, 0.f) * w2n;
        #pragma unroll
        for (int off = 1; off < 16; off <<= 1)
            v += __shfl_xor(v, off, 32);   // stays within 16-lane half
        red[r8] = v;
    }
    if (lm == 0) {
        #pragma unroll
        for (int r8 = 0; r8 < 8; ++r8) {
            const int row = tile * 16 + 8 * half + r8;
            const float sp = hk_softplus(red[r8] + b2v);
            if (role < 6) {
                tab[row] = sp;                       // 48 rows exactly
            } else if (half == 0 && r8 < 4) {
                tab[r8] = sp;                        // mu: rows 0..3
            }
        }
    }
}

// ---------------------------------------------------------------------------
// Kernel 3: fill mu / alpha / delta outputs (broadcast + table gather).
// ---------------------------------------------------------------------------
__global__ void hk_fill_kernel(const int* __restrict__ event,
                               const int* __restrict__ mkt,
                               const float* __restrict__ a_tab,
                               const float* __restrict__ d_tab,
                               const float* __restrict__ mu_base,
                               float* __restrict__ out) {
    const int idx = blockIdx.x * 256 + threadIdx.x;       // < 32768
    if (idx < Bsz * Tsz * 4)
        out[OFF_MU + idx] = mu_base[idx & 3];
    if (idx < Bsz * Isz * 4) {
        const int k  = idx & 3;
        const int bi = idx >> 2;            // b*1023 + i
        const int b  = bi / Isz;
        const int i  = bi - b * Isz;
        const int ev = event[b * Tsz + i];
        const int st = mkt[b * Tsz + i];
        const int c  = ev * 12 + st * 4 + k;
        out[OFF_ALPHA + idx] = a_tab[c];
        out[OFF_DELTA + idx] = d_tab[c];
    }
}

// ---------------------------------------------------------------------------
// Kernel 4: excitation. Block = 256 threads = 64 j values x 4 k.
// Grid = B * 16. i-dimension tiled through LDS; alpha/delta packed float2 ->
// one ds_load_b64 per term; full tiles unrolled x4 to pipeline DS vs v_exp.
// ---------------------------------------------------------------------------
__global__ void hk_excite_kernel(const float* __restrict__ ts,
                                 const float* __restrict__ alpha_o,
                                 const float* __restrict__ delta_o,
                                 const float* __restrict__ mu_base,
                                 float* __restrict__ intens) {
    __shared__ float  s_ts[TILE_I];
    __shared__ float2 s_ad[TILE_I * 4];    // (alpha, delta) pairs, 4 k per i

    const int b  = blockIdx.x >> 4;          // 16 j-tiles per batch
    const int jt = blockIdx.x & 15;
    const int t  = threadIdx.x;              // 0..255
    const int jj = t >> 2;
    const int k  = t & 3;
    const int j  = jt * 64 + jj;             // 0..1023
    const bool valid = (j < Isz);

    const float tsj = ts[b * Tsz + (valid ? j + 1 : Isz)];
    float acc = 0.f;

    const int imax_block = min(jt * 64 + 63, Isz - 1);
    for (int i0 = 0; i0 <= imax_block; i0 += TILE_I) {
        const int nload = min(TILE_I, Isz - i0);
        if (t < TILE_I)
            s_ts[t] = (t < nload) ? ts[b * Tsz + i0 + t] : 0.f;
        #pragma unroll 2
        for (int q = t; q < TILE_I * 4; q += 256) {
            const int ii = q >> 2;
            const int kk = q & 3;
            float av = 0.f, dv = 1.f;
            if (ii < nload) {
                const int p = (b * Isz + i0 + ii) * 4 + kk;
                av = alpha_o[p];
                dv = delta_o[p];
            }
            s_ad[q] = make_float2(av, dv);
        }
        // prefetch next ts tile (global_prefetch_b8)
        if (t == 0 && i0 + TILE_I <= imax_block)
            __builtin_prefetch(&ts[b * Tsz + i0 + TILE_I], 0, 0);
        __syncthreads();

        const int lim = min(nload, j - i0 + 1);   // i <= j
        if (valid && lim > 0) {
            if (lim == TILE_I) {
                // full tile: no bound check, unrolled for DS/TRANS pipelining
                #pragma unroll 4
                for (int ii = 0; ii < TILE_I; ++ii) {
                    const float2 ad = s_ad[ii * 4 + k];
                    float dtv = tsj - s_ts[ii];
                    dtv = (dtv > 0.f) ? dtv : 1.0e9f;
                    acc = fmaf(ad.x, __expf(-ad.y * dtv), acc);
                }
            } else {
                for (int ii = 0; ii < lim; ++ii) {
                    const float2 ad = s_ad[ii * 4 + k];
                    float dtv = tsj - s_ts[ii];
                    dtv = (dtv > 0.f) ? dtv : 1.0e9f;  // where(dt>0, dt, BIG)
                    acc = fmaf(ad.x, __expf(-ad.y * dtv), acc);
                }
            }
        }
        __syncthreads();
    }
    if (valid)
        intens[(b * Isz + j) * 4 + k] = mu_base[k] + acc;
}

// ---------------------------------------------------------------------------
extern "C" void kernel_launch(void* const* d_in, const int* in_sizes, int n_in,
                              void* d_out, int out_size, void* d_ws, size_t ws_size,
                              hipStream_t stream) {
    const float* time_step = (const float*)d_in[0];
    const int*   event     = (const int*)  d_in[1];
    const int*   mkt       = (const int*)  d_in[2];
    const float* mu_w1 = (const float*)d_in[3];
    const float* mu_b1 = (const float*)d_in[4];
    const float* mu_w2 = (const float*)d_in[5];
    const float* mu_b2 = (const float*)d_in[6];
    const float* a_w1  = (const float*)d_in[7];
    const float* a_b1  = (const float*)d_in[8];
    const float* a_w2  = (const float*)d_in[9];
    const float* a_b2  = (const float*)d_in[10];
    const float* d_w1  = (const float*)d_in[11];
    const float* d_b1  = (const float*)d_in[12];
    const float* d_w2  = (const float*)d_in[13];
    const float* d_b2  = (const float*)d_in[14];

    float* out = (float*)d_out;
    float* ws  = (float*)d_ws;
    float* ts      = ws;           // 8192 floats
    float* a_tab   = ws + 8192;    // 48
    float* d_tab   = ws + 8240;    // 48
    float* mu_base = ws + 8288;    // 4

    hk_cumsum_kernel<<<1, 256, 0, stream>>>(time_step, ts);
    hk_tables_kernel<<<7, 32, 0, stream>>>(mu_w1, mu_b1, mu_w2, mu_b2,
                                           a_w1, a_b1, a_w2, a_b2,
                                           d_w1, d_b1, d_w2, d_b2,
                                           a_tab, d_tab, mu_base);
    hk_fill_kernel<<<128, 256, 0, stream>>>(event, mkt, a_tab, d_tab, mu_base, out);
    hk_excite_kernel<<<Bsz * 16, 256, 0, stream>>>(ts, out + OFF_ALPHA, out + OFF_DELTA,
                                                   mu_base, out);
}
